// GRUSummary_14937896255804
// MI455X (gfx1250) — matrix operand order repaired and checked by reading, not compiled, not run on hardware
//
#include <hip/hip_runtime.h>
#include <hip/hip_bf16.h>

typedef __attribute__((ext_vector_type(16))) __bf16 v16bf;
typedef __attribute__((ext_vector_type(8)))  float  v8f;

#define BB   512
#define TT   1024
#define UU   256
#define NU   768      // 3*U
#define SS   16
#define BM   16       // batch rows per workgroup
#define NTILES 48     // NU/16
#define TPW  6        // n-tiles per wave (48 / 8 waves)
#define KST  8        // K steps: 256 / 32
#define FRAG_ELEMS (32*16)   // one (tile,kstep) fragment: 32 lanes x 16 bf16

// ---------------------------------------------------------------------------
// Prep: U (256x768 f32, row-major KxN) -> bf16, pre-swizzled into WMMA
// B-fragment order: wfrag[((ntile*8 + kstep)*32 + lane)*16 + e]
//   n = ntile*16 + (lane&15)
//   k = kstep*32 + ((lane>>4)<<4) + e      (contiguous k order; A frags use the same)
// Each wave then loads its fragment as one contiguous 1KB burst.
// ---------------------------------------------------------------------------
__global__ void prep_wfrag(const float* __restrict__ Um, __bf16* __restrict__ wfrag) {
    int idx = blockIdx.x * blockDim.x + threadIdx.x;   // [0, 48*8*32*16)
    if (idx >= NTILES * KST * FRAG_ELEMS) return;
    int e    = idx & 15;
    int lane = (idx >> 4) & 31;
    int s    = (idx >> 9) & 7;
    int nt   = idx >> 12;
    int k = s * 32 + ((lane >> 4) << 4) + e;
    int n = nt * 16 + (lane & 15);
    wfrag[idx] = (__bf16)Um[k * NU + n];
}

// ---------------------------------------------------------------------------
// Persistent GRU: one workgroup = 16 batch rows, 8 wave32s, full T loop.
// Per step: rec = h @ U via v_wmma_f32_16x16x32_bf16 (A=h from LDS frag
// layout, B=wfrag from L2), rec staged in LDS, gate math elementwise,
// h updated in LDS (bf16, A-fragment layout). Epilogue: LN + dense.
// ---------------------------------------------------------------------------
__global__ __launch_bounds__(256) void gru_persistent(
    const float* __restrict__ x,      // (B, T)
    const float* __restrict__ W,      // (1, 768)
    const __bf16* __restrict__ wfrag, // swizzled bf16 weights
    const float* __restrict__ b_i,    // (768)
    const float* __restrict__ b_r,    // (768)
    const float* __restrict__ gam,    // (256)
    const float* __restrict__ bet,    // (256)
    const float* __restrict__ Wd,     // (256,16)
    const float* __restrict__ bd,     // (16)
    float* __restrict__ out)          // (B,16)
{
    __shared__ float s_rec[BM][NU];                       // 48 KB
    __shared__ __align__(32) __bf16 s_hfrag[KST * FRAG_ELEMS]; // 8 KB, A-frag layout
    __shared__ float s_W[NU];                             // 3 KB
    __shared__ float s_bsum[512];                         // b_i+b_r for z,r gates
    __shared__ float s_bih[256];                          // b_i h-gate
    __shared__ float s_brh[256];                          // b_r h-gate
    __shared__ float s_x[BM];
    __shared__ float s_mean[BM];
    __shared__ float s_rstd[BM];

    const int tid  = threadIdx.x;
    const int lane = tid & 31;          // wave32
    const int wv   = tid >> 5;          // 8 waves
    const int r0   = blockIdx.x * BM;

    // ---- init ----
    for (int i = tid; i < KST * FRAG_ELEMS; i += 256) s_hfrag[i] = (__bf16)0.0f;
    for (int i = tid; i < NU; i += 256) s_W[i] = W[i];
    for (int i = tid; i < 512; i += 256) s_bsum[i] = b_i[i] + b_r[i];
    s_bih[tid] = b_i[512 + tid];
    s_brh[tid] = b_r[512 + tid];
    __syncthreads();

    // per-thread gate constants: this thread owns hidden unit j = tid
    const int j    = tid;
    const int js   = j >> 5;
    const int jo   = j & 31;
    const int loff = ((jo >> 4) & 1) << 4;              // +16 lanes for K 16..31
    const int je   = jo & 15;                           // element within v16bf
    const float wj0 = s_W[j];
    const float wj1 = s_W[j + 256];
    const float wj2 = s_W[j + 512];
    const float bs0 = s_bsum[j];
    const float bs1 = s_bsum[j + 256];
    const float bih = s_bih[j];
    const float brh = s_brh[j];

    const __bf16* wb = wfrag + (size_t)(wv * TPW) * (KST * FRAG_ELEMS);
    const int colb = lane & 15;
    const int rowb = (lane & 16) ? 8 : 0;

    for (int t = 0; t < TT; ++t) {
        __syncthreads();  // prev gate writes to s_hfrag visible; prev s_rec reads done
        if (tid < BM) s_x[tid] = x[(size_t)(r0 + tid) * TT + t];

        // ---- WMMA phase: rec = h @ U ----
        v8f acc[TPW];
        #pragma unroll
        for (int nt = 0; nt < TPW; ++nt) acc[nt] = (v8f){0.f,0.f,0.f,0.f,0.f,0.f,0.f,0.f};

        #pragma unroll
        for (int s = 0; s < KST; ++s) {
            v16bf a = *(const v16bf*)&s_hfrag[(s * 32 + lane) * 16];
            #pragma unroll
            for (int nt = 0; nt < TPW; ++nt) {
                v16bf b = *(const v16bf*)(wb + ((nt * KST + s) * 32 + lane) * 16);
                acc[nt] = __builtin_amdgcn_wmma_f32_16x16x32_bf16(
                    false, a, false, b, (short)0, acc[nt], false, false);
            }
        }

        // scatter accumulators -> s_rec (C layout: vgpr d, lanes<16: M=d, else M=d+8)
        #pragma unroll
        for (int nt = 0; nt < TPW; ++nt) {
            int col = (wv * TPW + nt) * 16 + colb;
            #pragma unroll
            for (int d = 0; d < 8; ++d) s_rec[rowb + d][col] = acc[nt][d];
        }
        __syncthreads();  // rec + s_x ready

        // ---- gate phase: thread owns column j for all 16 rows ----
        #pragma unroll 4
        for (int row = 0; row < BM; ++row) {
            float xv = s_x[row];
            float rz = s_rec[row][j];
            float rr = s_rec[row][j + 256];
            float rh = s_rec[row][j + 512];
            float z  = 1.0f / (1.0f + expf(-(fmaf(xv, wj0, bs0) + rz)));
            float r  = 1.0f / (1.0f + expf(-(fmaf(xv, wj1, bs1) + rr)));
            float hh = tanhf(fmaf(xv, wj2, bih) + r * (rh + brh));
            int fa = (js * 32 + row + loff) * 16 + je;
            float hold = (float)s_hfrag[fa];
            s_hfrag[fa] = (__bf16)(z * hold + (1.0f - z) * hh);
        }
    }
    __syncthreads();

    // ---- epilogue: LayerNorm + dense (256 -> 16) ----
    for (int row = 0; row < BM; ++row) {
        int fa = (js * 32 + row + loff) * 16 + je;
        s_rec[row][j] = (float)s_hfrag[fa];      // h as f32, reuse s_rec
    }
    __syncthreads();
    if (tid < BM) {
        float sum = 0.f, ssq = 0.f;
        for (int k = 0; k < UU; ++k) { float v = s_rec[tid][k]; sum += v; ssq += v * v; }
        float mean = sum * (1.0f / 256.0f);
        float var  = ssq * (1.0f / 256.0f) - mean * mean;
        s_mean[tid] = mean;
        s_rstd[tid] = rsqrtf(var + 1e-3f);
    }
    __syncthreads();
    {
        int row = tid >> 4, sc = tid & 15;
        float mean = s_mean[row], rstd = s_rstd[row];
        float a = bd[sc];
        for (int k = 0; k < UU; ++k) {
            float hn = (s_rec[row][k] - mean) * rstd * gam[k] + bet[k];
            a = fmaf(hn, Wd[k * SS + sc], a);
        }
        out[(size_t)(r0 + row) * SS + sc] = a;
    }
}

// ---------------------------------------------------------------------------
extern "C" void kernel_launch(void* const* d_in, const int* in_sizes, int n_in,
                              void* d_out, int out_size, void* d_ws, size_t ws_size,
                              hipStream_t stream) {
    const float* x   = (const float*)d_in[0];  // time_series (512,1024,1)
    const float* W   = (const float*)d_in[1];  // (1,768)
    const float* Um  = (const float*)d_in[2];  // (256,768)
    const float* bi  = (const float*)d_in[3];
    const float* br  = (const float*)d_in[4];
    const float* gam = (const float*)d_in[5];
    const float* bet = (const float*)d_in[6];
    const float* Wd  = (const float*)d_in[7];
    const float* bd  = (const float*)d_in[8];
    float* out = (float*)d_out;
    __bf16* wfrag = (__bf16*)d_ws;             // 48*8*32*16*2 = 393216 B

    prep_wfrag<<<768, 256, 0, stream>>>(Um, wfrag);
    gru_persistent<<<BB / BM, 256, 0, stream>>>(x, W, wfrag, bi, br, gam, bet, Wd, bd, out);
}
